// SwinOperation_4561255268471
// MI455X (gfx1250) — matrix operand order repaired and checked by reading, not compiled, not run on hardware
//
#include <hip/hip_runtime.h>
#include <hip/hip_bf16.h>

// ---------------- problem constants ----------------
#define WSZ     7
#define SHIFT_  3
#define HEADS_  3
#define HD_     32
#define CH_     96      // channels
#define IMG_    56      // feature map H=W
#define NWD_    8       // windows per dim
#define NWIN_I  64      // windows per image
#define NTOK    49      // tokens per window
#define NPAD    64      // padded tokens (multiple of 16)
#define BATCH_  128
#define TOTWIN  (BATCH_ * NWIN_I)   // 8192

typedef __attribute__((ext_vector_type(16))) _Float16 v16h;
typedef __attribute__((ext_vector_type(8)))  _Float16 v8h;
typedef __attribute__((ext_vector_type(8)))  float    v8f;

static __device__ __forceinline__ v16h cat16(v8h lo, v8h hi) {
    v16h r;
#pragma unroll
    for (int i = 0; i < 8; i++) { r[i] = lo[i]; r[8 + i] = hi[i]; }
    return r;
}

// A fragment (16x32 f16, MxK) from row-major f16, row stride S (elements).
// Lanes 0-15: row m, K {0..7, 16..23}; lanes 16-31: row m, K {8..15, 24..31}.
static __device__ __forceinline__ v16h load_a(const _Float16* p, int S, int r0, int k0, int lane) {
    const int m  = lane & 15;
    const int kh = (lane >> 4) << 3;          // 0 or 8
    const _Float16* base = p + (size_t)(r0 + m) * S + k0 + kh;
    v8h lo = *(const v8h*)(base);             // K = kh .. kh+7
    v8h hi = *(const v8h*)(base + 16);        // K = 16+kh .. 16+kh+7
    return cat16(lo, hi);
}

// B fragment (32x16 f16, KxN) where B[k][n] = M[n0+n][k0+k] (M row-major, stride S).
// Lanes 0-15 hold K {0..7, 16..23} of column n; lanes 16-31 hold K {8..15, 24..31}.
static __device__ __forceinline__ v16h load_bT(const _Float16* p, int S, int n0, int k0, int lane) {
    const int n  = lane & 15;
    const int kh = (lane >> 4) << 3;          // 0 or 8
    const _Float16* base = p + (size_t)(n0 + n) * S + k0 + kh;
    v8h lo = *(const v8h*)(base);
    v8h hi = *(const v8h*)(base + 16);
    return cat16(lo, hi);
}

static __device__ __forceinline__ v8f wmma_f16(v16h a, v16h b, v8f c) {
    return __builtin_amdgcn_wmma_f32_16x16x32_f16(false, a, false, b, (short)0, c, false, false);
}

static __device__ __forceinline__ int region3(int p) {  // shift-mask region id
    return (p < IMG_ - WSZ) ? 0 : ((p < IMG_ - SHIFT_) ? 1 : 2);
}

// ---------------- kernel 0: weights fp32 -> f16 ----------------------------
__global__ void k_cvt_w(const float* __restrict__ qw, const float* __restrict__ pw,
                        _Float16* __restrict__ qwh, _Float16* __restrict__ pwh) {
    int gid = blockIdx.x * blockDim.x + threadIdx.x;
    const int NQ = 3 * CH_ * CH_, NP = CH_ * CH_;
    if (gid < NQ)            qwh[gid]      = (_Float16)qw[gid];
    else if (gid < NQ + NP)  pwh[gid - NQ] = (_Float16)pw[gid - NQ];
}

// ---------------- kernel 1: fully fused Swin window attention --------------
// One block (4 wave32s) per window. LayerNorm + shift + partition are fused
// into stage 0 (direct gather from x), so x and out are each touched once.
__launch_bounds__(128, 1)
__global__ void k_win_attn(const float*    __restrict__ x,
                           const float*    __restrict__ lng,
                           const float*    __restrict__ lnb,
                           const _Float16* __restrict__ qwh,
                           const float*    __restrict__ qkvb,
                           const _Float16* __restrict__ pwh,
                           const float*    __restrict__ pjb,
                           const float*    __restrict__ rel,
                           float*          __restrict__ out) {
    __shared__ _Float16 sX[NPAD * CH_];        // 12 KB  (LN'd window; reused for O)
    __shared__ _Float16 sQK[NPAD * 2 * CH_];   // 24 KB  (cols 0..95 = Q*scale, 96..191 = K)
    __shared__ _Float16 sVT[CH_ * NPAD];       // 12 KB  (V transposed: [d][token])
    __shared__ _Float16 sP[NPAD * NPAD];       //  8 KB  (softmax probs, f16)

    const int tid  = threadIdx.x;
    const int lane = tid & 31;                 // wave32
    const int wid  = tid >> 5;                 // 4 waves / block
    const int win  = blockIdx.x;
    const int bi   = win >> 6;
    const int widx = win & 63;
    const int wy = widx >> 3, wx = widx & 7;
    const int m0   = wid * 16;                 // this wave's M-tile rows
    const int nloc = lane & 15;
    const int rb8  = (lane >> 4) << 3;         // D-layout: lanes16-31 hold rows r+8

    __builtin_prefetch(qwh, 0, 0);
    __builtin_prefetch(pwh, 0, 0);

    // ---- stage 0: fused LayerNorm + roll(-3,-3) + partition + f16 pack
    // Two threads per token: each handles 48 channels, partner-sum via shfl.
    {
        const int tok  = tid >> 1;             // 0..63
        const int half = tid & 1;              // channel half
        if (tok < NTOK) {
            const int py = tok / WSZ, px = tok % WSZ;
            const int hh = (wy * WSZ + py + SHIFT_) % IMG_;   // rolled coord -> source coord
            const int ww = (wx * WSZ + px + SHIFT_) % IMG_;
            const float* xp = x + (size_t)bi * CH_ * IMG_ * IMG_
                                + (size_t)half * 48 * IMG_ * IMG_
                                + hh * IMG_ + ww;
            float v[48];
            float s = 0.f, s2 = 0.f;
#pragma unroll
            for (int c = 0; c < 48; c++) {
                float t = xp[(size_t)c * IMG_ * IMG_];
                v[c] = t; s += t; s2 += t * t;
            }
            s  += __shfl_xor(s, 1);            // partner thread holds other 48 channels
            s2 += __shfl_xor(s2, 1);
            const float mu  = s  * (1.0f / CH_);
            const float var = s2 * (1.0f / CH_) - mu * mu;
            const float rs  = rsqrtf(var + 1e-5f);
            _Float16* dst = sX + tok * CH_ + half * 48;
#pragma unroll
            for (int c = 0; c < 48; c++)
                dst[c] = (_Float16)((v[c] - mu) * rs * lng[half * 48 + c] + lnb[half * 48 + c]);
        } else {
            // padded rows 49..63 -> zeros
            v8h z = {};
            _Float16* dst = sX + tok * CH_ + half * 48;
#pragma unroll
            for (int c8 = 0; c8 < 6; c8++)
                *(v8h*)(dst + c8 * 8) = z;
        }
    }
    __syncthreads();

    // ---- stage 1: QKV = X @ W^T + b  (64x288, K=96), WMMA f16
    const float scale = 0.17677669529663687f;  // hd^-0.5
    {
        v16h a0 = load_a(sX, CH_, m0, 0,  lane);
        v16h a1 = load_a(sX, CH_, m0, 32, lane);
        v16h a2 = load_a(sX, CH_, m0, 64, lane);
        for (int nt = 0; nt < 18; nt++) {
            const int n0 = nt * 16;
            v16h b0 = load_bT(qwh, CH_, n0, 0,  lane);
            v16h b1 = load_bT(qwh, CH_, n0, 32, lane);
            v16h b2 = load_bT(qwh, CH_, n0, 64, lane);
            v8f acc = {};
            acc = wmma_f16(a0, b0, acc);
            acc = wmma_f16(a1, b1, acc);
            acc = wmma_f16(a2, b2, acc);
            const int col = n0 + nloc;
            const float bb = qkvb[col];
#pragma unroll
            for (int r = 0; r < 8; r++) {
                float vv = acc[r] + bb;
                int row = m0 + rb8 + r;
                if (col < CH_)            sQK[row * (2 * CH_) + col] = (_Float16)(vv * scale);
                else if (col < 2 * CH_)   sQK[row * (2 * CH_) + col] = (_Float16)vv;
                else                      sVT[(col - 2 * CH_) * NPAD + row] = (_Float16)vv;
            }
        }
    }
    __syncthreads();

    // ---- stage 2: per-head attention, all LDS-resident, no cross-wave deps
    for (int h = 0; h < HEADS_; h++) {
        // S = Qs @ K^T : wave owns rows m0..m0+15, 4 column tiles
        v16h qa = load_a(sQK, 2 * CH_, m0, h * HD_, lane);
        v8f sc[4];
#pragma unroll
        for (int jt = 0; jt < 4; jt++) {
            v16h kb = load_bT(sQK, 2 * CH_, jt * 16, CH_ + h * HD_, lane);
            v8f z = {};
            sc[jt] = wmma_f16(qa, kb, z);
        }
        // logits = S + rel_bias + shift_mask (computed analytically)
        float lg[4][8];
#pragma unroll
        for (int jt = 0; jt < 4; jt++) {
            const int j = jt * 16 + nloc;
            const int rj = j / WSZ, cj = j % WSZ;
            int cntj = 0;
            if (j < NTOK)
                cntj = region3(wy * WSZ + rj) * 3 + region3(wx * WSZ + cj);
#pragma unroll
            for (int r = 0; r < 8; r++) {
                const int i = m0 + rb8 + r;
                float v;
                if (j >= NTOK)      v = -1e9f;     // mask padded keys
                else if (i >= NTOK) v = 0.0f;      // padded query rows: benign
                else {
                    const int ri = i / WSZ, ci = i % WSZ;
                    const int cnti = region3(wy * WSZ + ri) * 3 + region3(wx * WSZ + ci);
                    const int idx = (ri - rj + WSZ - 1) * (2 * WSZ - 1) + (ci - cj + WSZ - 1);
                    v = sc[jt][r] + rel[idx * HEADS_ + h];
                    if (cnti != cntj) v -= 100.0f;
                }
                lg[jt][r] = v;
            }
        }
        // softmax per row; row i lives in one 16-lane half, so xor(1,2,4,8) reduces it
#pragma unroll
        for (int r = 0; r < 8; r++) {
            float mx = fmaxf(fmaxf(lg[0][r], lg[1][r]), fmaxf(lg[2][r], lg[3][r]));
            mx = fmaxf(mx, __shfl_xor(mx, 1));
            mx = fmaxf(mx, __shfl_xor(mx, 2));
            mx = fmaxf(mx, __shfl_xor(mx, 4));
            mx = fmaxf(mx, __shfl_xor(mx, 8));
            float sum = 0.f;
#pragma unroll
            for (int jt = 0; jt < 4; jt++) { lg[jt][r] = __expf(lg[jt][r] - mx); sum += lg[jt][r]; }
            sum += __shfl_xor(sum, 1);
            sum += __shfl_xor(sum, 2);
            sum += __shfl_xor(sum, 4);
            sum += __shfl_xor(sum, 8);
            const float inv = 1.0f / sum;
            const int i = m0 + rb8 + r;
#pragma unroll
            for (int jt = 0; jt < 4; jt++)
                sP[i * NPAD + jt * 16 + nloc] = (_Float16)(lg[jt][r] * inv);
        }
        // O_head = P @ V  (64x32, K=64); own-wave rows only -> no barrier needed
        v16h pa0 = load_a(sP, NPAD, m0, 0,  lane);
        v16h pa1 = load_a(sP, NPAD, m0, 32, lane);
#pragma unroll
        for (int nt = 0; nt < 2; nt++) {
            v16h vb0 = load_bT(sVT, NPAD, h * HD_ + nt * 16, 0,  lane);
            v16h vb1 = load_bT(sVT, NPAD, h * HD_ + nt * 16, 32, lane);
            v8f o = {};
            o = wmma_f16(pa0, vb0, o);
            o = wmma_f16(pa1, vb1, o);
            const int d = h * HD_ + nt * 16 + nloc;
#pragma unroll
            for (int r = 0; r < 8; r++)
                sX[(m0 + rb8 + r) * CH_ + d] = (_Float16)o[r];   // sX reused as O
        }
    }
    __syncthreads();

    // ---- stage 3: proj GEMM + fused window-reverse + unshift + BCHW scatter
    {
        v16h oa0 = load_a(sX, CH_, m0, 0,  lane);
        v16h oa1 = load_a(sX, CH_, m0, 32, lane);
        v16h oa2 = load_a(sX, CH_, m0, 64, lane);
        for (int nt = 0; nt < 6; nt++) {
            const int n0 = nt * 16;
            v16h b0 = load_bT(pwh, CH_, n0, 0,  lane);
            v16h b1 = load_bT(pwh, CH_, n0, 32, lane);
            v16h b2 = load_bT(pwh, CH_, n0, 64, lane);
            v8f acc = {};
            acc = wmma_f16(oa0, b0, acc);
            acc = wmma_f16(oa1, b1, acc);
            acc = wmma_f16(oa2, b2, acc);
            const int c = n0 + nloc;
            const float bb = pjb[c];
#pragma unroll
            for (int r = 0; r < 8; r++) {
                const int i = m0 + rb8 + r;
                if (i < NTOK) {
                    const int py = i / WSZ, px = i % WSZ;
                    const int hh = (wy * WSZ + py + SHIFT_) % IMG_;  // roll(+3)
                    const int ww = (wx * WSZ + px + SHIFT_) % IMG_;
                    out[(((size_t)bi * CH_ + c) * IMG_ + hh) * IMG_ + ww] = acc[r] + bb;
                }
            }
        }
    }
}

// ---------------- host-side launch ----------------
extern "C" void kernel_launch(void* const* d_in, const int* in_sizes, int n_in,
                              void* d_out, int out_size, void* d_ws, size_t ws_size,
                              hipStream_t stream) {
    const float* x      = (const float*)d_in[0];
    const float* ln_g   = (const float*)d_in[1];
    const float* ln_b   = (const float*)d_in[2];
    const float* qkv_w  = (const float*)d_in[3];
    const float* qkv_b  = (const float*)d_in[4];
    const float* proj_w = (const float*)d_in[5];
    const float* proj_b = (const float*)d_in[6];
    const float* rel    = (const float*)d_in[7];
    float* out = (float*)d_out;

    _Float16* ws  = (_Float16*)d_ws;
    _Float16* qwh = ws;                       // [288][96] f16
    _Float16* pwh = qwh + 3 * CH_ * CH_;      // [96][96]  f16  (total ~74 KB of ws)

    {
        int total = 3 * CH_ * CH_ + CH_ * CH_;
        k_cvt_w<<<(total + 255) / 256, 256, 0, stream>>>(qkv_w, proj_w, qwh, pwh);
    }
    k_win_attn<<<TOTWIN, 128, 0, stream>>>(x, ln_g, ln_b, qwh, qkv_b, pwh, proj_b, rel, out);
}
